// MambaFuser_82325933130035
// MI455X (gfx1250) — compile-verified, matrix operand order checked by reading, not run on hardware
//
#include <hip/hip_runtime.h>
#include <hip/hip_bf16.h>

typedef __attribute__((ext_vector_type(16))) _Float16 v16h;
typedef __attribute__((ext_vector_type(8)))  float    v8f;

#define DMODEL 256
#define DINNER 512
#define DSTATE 16
#define DTRANK 16
#define LSEQ   4096
#define NBATCH 8
#define ROWS   (NBATCH*LSEQ)   // 32768

__device__ __forceinline__ float sigmoidf_(float x){ return 1.f/(1.f+__expf(-x)); }
__device__ __forceinline__ float siluf_(float x){ return x*sigmoidf_(x); }
__device__ __forceinline__ float softplusf_(float x){ return (x>20.f)?x:__logf(1.f+__expf(x)); }

// ---------------- spiral gather: stack_feats -> x [8,4096,256] ----------------
__global__ __launch_bounds__(256) void k_gather(const float* __restrict__ stack,
                                                const int* __restrict__ co,
                                                float* __restrict__ xbuf)
{
  int idx = blockIdx.x*256 + threadIdx.x;          // 2^23 total
  int d = idx & 255;
  int l = (idx >> 8) & 4095;
  int b = idx >> 20;
  int p = co[l];
  int bb = (d < 128) ? b : (b + 8);
  int cc = d & 127;
  xbuf[idx] = stack[((size_t)(bb*128 + cc) << 12) + p];
}

// ---------------- layernorm over 256, one wave per row ----------------
__global__ __launch_bounds__(256) void k_ln(const float* __restrict__ X,
                                            const float* __restrict__ w,
                                            const float* __restrict__ b,
                                            float* __restrict__ Y)
{
  int wid  = (blockIdx.x * 256 + threadIdx.x) >> 5;
  int lane = threadIdx.x & 31;
  const float* r = X + (size_t)wid * DMODEL;
  float v[8]; float s = 0.f;
  #pragma unroll
  for (int i=0;i<8;i++){ v[i] = r[i*32 + lane]; s += v[i]; }
  #pragma unroll
  for (int m=16;m;m>>=1) s += __shfl_xor(s, m, 32);
  float mu = s * (1.f/DMODEL);
  float q = 0.f;
  #pragma unroll
  for (int i=0;i<8;i++){ float d2 = v[i]-mu; q += d2*d2; }
  #pragma unroll
  for (int m=16;m;m>>=1) q += __shfl_xor(q, m, 32);
  float rs = rsqrtf(q*(1.f/DMODEL) + 1e-5f);
  float* o = Y + (size_t)wid * DMODEL;
  #pragma unroll
  for (int i=0;i<8;i++){
    int c = i*32 + lane;
    o[c] = (v[i]-mu)*rs*w[c] + b[c];
  }
}

// ---------------- final layernorm + spiral scatter to output ----------------
__global__ __launch_bounds__(256) void k_lnf_scatter(const float* __restrict__ X,
    const float* __restrict__ w, const float* __restrict__ b,
    const int* __restrict__ co, float* __restrict__ out)
{
  int wid  = (blockIdx.x * 256 + threadIdx.x) >> 5;
  int lane = threadIdx.x & 31;
  int bb = wid >> 12;
  int l  = wid & 4095;
  const float* r = X + (size_t)wid * DMODEL;
  float v[8]; float s = 0.f;
  #pragma unroll
  for (int i=0;i<8;i++){ v[i] = r[i*32 + lane]; s += v[i]; }
  #pragma unroll
  for (int m=16;m;m>>=1) s += __shfl_xor(s, m, 32);
  float mu = s * (1.f/DMODEL);
  float q = 0.f;
  #pragma unroll
  for (int i=0;i<8;i++){ float d2 = v[i]-mu; q += d2*d2; }
  #pragma unroll
  for (int m=16;m;m>>=1) q += __shfl_xor(q, m, 32);
  float rs = rsqrtf(q*(1.f/DMODEL) + 1e-5f);
  int p = co[l];
  #pragma unroll
  for (int i=0;i<8;i++){
    int c = i*32 + lane;
    out[((size_t)(bb*256 + c) << 12) + p] = (v[i]-mu)*rs*w[c] + b[c];
  }
}

// ---------------- causal depthwise conv (K=4) + SiLU ----------------
// reads u from xz columns [0,512) (ld=1024), writes uc (ld=512)
__global__ __launch_bounds__(256) void k_conv(const float* __restrict__ xz,
  const float* __restrict__ cw, const float* __restrict__ cb,
  float* __restrict__ uc)
{
  int idx = blockIdx.x*256 + threadIdx.x;          // 2^24 total
  int d = idx & 511;
  int t = (idx >> 9) & 4095;
  int b = idx >> 21;
  size_t row = (size_t)(b*LSEQ + t);
  const float* up = xz + row*1024 + d;
  const float* w  = cw + d*4;
  float a = cb[d];
  #pragma unroll
  for (int j=0;j<4;j++){
    int tt = t - 3 + j;
    if (tt >= 0) a += w[j] * up[(ptrdiff_t)(j-3)*1024];
  }
  uc[row*512 + d] = siluf_(a);
}

// ---------------- selective scan: one lane per (batch, channel) ----------------
// reads uc, dt, B/C (xdbl cols 16..47), z (xz cols 512..); writes gated y into xz cols [0,512)
__global__ __launch_bounds__(256) void k_scan(
  const float* __restrict__ uc, const float* __restrict__ dtb,
  const float* __restrict__ xdbl, float* __restrict__ xz,
  const float* __restrict__ Alog, const float* __restrict__ Dp)
{
  int tid = blockIdx.x*256 + threadIdx.x;          // 4096 lanes
  int b = tid >> 9;
  int d = tid & 511;
  float Am[DSTATE];
  #pragma unroll
  for (int s=0;s<DSTATE;s++) Am[s] = -__expf(Alog[(size_t)d*DSTATE + s]);
  float Dd = Dp[d];
  float h[DSTATE];
  #pragma unroll
  for (int s=0;s<DSTATE;s++) h[s] = 0.f;
  size_t rbase = (size_t)b * LSEQ;
  for (int t=0;t<LSEQ;t++){
    size_t row = rbase + t;
    float dt = dtb[row*512 + d];
    float u  = uc [row*512 + d];
    const float* bc = xdbl + row*48 + DTRANK;      // B[0..15], C[16..31]
    float du = dt * u;
    float y = 0.f;
    #pragma unroll
    for (int s=0;s<DSTATE;s++){
      h[s] = __expf(dt*Am[s])*h[s] + du*bc[s];
      y += h[s]*bc[DSTATE+s];
    }
    float zv = xz[row*1024 + 512 + d];
    y = (y + u*Dd) * siluf_(zv);
    xz[row*1024 + d] = y;
  }
}

// ---------------- f16 WMMA GEMM: Out[M,N] = A[M,K] * W[N,K]^T ----------------
// A-operand (16x32, §7.12.2): lane<16 -> row=lane, k-chunks {0..7,16..23};
// lane>=16 -> same row, k-chunks {8..15,24..31}. B-operand mirrors with lane=N col.
template<bool G>
__device__ __forceinline__ v16h load_half_op(const float* __restrict__ rowp, int k0, int K)
{
  v16h r;
  #pragma unroll
  for (int i=0;i<8;i++){
    float a = (!G || (k0+i)    < K) ? rowp[k0+i]    : 0.f;
    float c = (!G || (k0+16+i) < K) ? rowp[k0+16+i] : 0.f;
    r[i]   = (_Float16)a;
    r[i+8] = (_Float16)c;
  }
  return r;
}

// EPI: 0 = store, 1 = softplus(acc + bias[n]) store, 2 = residual add
template<int MT, int NT, int EPI, bool KG>
__global__ __launch_bounds__(32) void k_gemm(
  const float* __restrict__ A, int lda,
  const float* __restrict__ Bw, int ldb,
  float* __restrict__ Out, int ldo,
  const float* __restrict__ bias,
  int K)
{
  int lane  = threadIdx.x;
  int rsel  = lane & 15;
  int koff  = (lane >> 4) << 3;
  int mbase = blockIdx.x * (16*MT);
  int nbase = blockIdx.y * (16*NT);

  v8f acc[MT][NT];
  #pragma unroll
  for (int i=0;i<MT;i++)
    #pragma unroll
    for (int j=0;j<NT;j++)
      #pragma unroll
      for (int v=0;v<8;v++) acc[i][j][v] = 0.f;

  const float* arow[MT];
  const float* brow[NT];
  #pragma unroll
  for (int i=0;i<MT;i++) arow[i] = A  + (size_t)(mbase + i*16 + rsel)*lda;
  #pragma unroll
  for (int j=0;j<NT;j++) brow[j] = Bw + (size_t)(nbase + j*16 + rsel)*ldb;

  for (int kb=0; kb<K; kb+=32){
    v16h av[MT], bv[NT];
    #pragma unroll
    for (int i=0;i<MT;i++) av[i] = load_half_op<KG>(arow[i], kb+koff, K);
    #pragma unroll
    for (int j=0;j<NT;j++) bv[j] = load_half_op<KG>(brow[j], kb+koff, K);
    if (kb + 32 < K){
      #pragma unroll
      for (int i=0;i<MT;i++) __builtin_prefetch(arow[i] + kb + 32 + koff, 0, 1);
    }
    #pragma unroll
    for (int i=0;i<MT;i++)
      #pragma unroll
      for (int j=0;j<NT;j++)
        acc[i][j] = __builtin_amdgcn_wmma_f32_16x16x32_f16(
            false, av[i], false, bv[j], (short)0, acc[i][j], false, false);
  }

  int mtop = 8*(lane>>4);
  #pragma unroll
  for (int i=0;i<MT;i++){
    #pragma unroll
    for (int j=0;j<NT;j++){
      int ncol = nbase + j*16 + rsel;
      float bval = (EPI==1) ? bias[ncol] : 0.f;
      #pragma unroll
      for (int v=0;v<8;v++){
        int mrow = mbase + i*16 + mtop + v;
        size_t o = (size_t)mrow*ldo + ncol;
        float r = acc[i][j][v];
        if (EPI==1) r = softplusf_(r + bval);
        if (EPI==2) Out[o] += r; else Out[o] = r;
      }
    }
  }
}

extern "C" void kernel_launch(void* const* d_in, const int* in_sizes, int n_in,
                              void* d_out, int out_size, void* d_ws, size_t ws_size,
                              hipStream_t stream)
{
  (void)in_sizes; (void)n_in; (void)out_size; (void)ws_size;
  const float* stack = (const float*)d_in[0];
  const float* ln_w  = (const float*)d_in[1];
  const float* ln_b  = (const float*)d_in[2];
  const float* Win   = (const float*)d_in[3];
  const float* convw = (const float*)d_in[4];
  const float* convb = (const float*)d_in[5];
  const float* Wx    = (const float*)d_in[6];
  const float* Wdt   = (const float*)d_in[7];
  const float* bdt   = (const float*)d_in[8];
  const float* Alog  = (const float*)d_in[9];
  const float* Dp    = (const float*)d_in[10];
  const float* Wout  = (const float*)d_in[11];
  const float* lnfw  = (const float*)d_in[12];
  const float* lnfb  = (const float*)d_in[13];
  const int*   co    = (const int*)d_in[14];
  float* out = (float*)d_out;

  // workspace layout (floats): x | xn | xz(u,z / later y,z) | uc | xdbl | dt
  float* xbuf = (float*)d_ws;                       // [32768,256]
  float* xn   = xbuf + (size_t)ROWS*DMODEL;         // [32768,256]
  float* xz   = xn   + (size_t)ROWS*DMODEL;         // [32768,1024]
  float* uc   = xz   + (size_t)ROWS*1024;           // [32768,512]
  float* xdbl = uc   + (size_t)ROWS*DINNER;         // [32768,48]
  float* dtb  = xdbl + (size_t)ROWS*48;             // [32768,512]

  k_gather<<<dim3((ROWS*DMODEL)/256), 256, 0, stream>>>(stack, co, xbuf);

  for (int i=0;i<4;i++){
    k_ln<<<dim3(ROWS/8), 256, 0, stream>>>(xbuf, ln_w + i*DMODEL, ln_b + i*DMODEL, xn);

    // in_proj: [32768,256] x [1024,256]^T -> xz [32768,1024]
    k_gemm<2,4,0,false><<<dim3(ROWS/32, 1024/64), 32, 0, stream>>>(
        xn, DMODEL, Win + (size_t)i*1024*DMODEL, DMODEL, xz, 1024, nullptr, DMODEL);

    k_conv<<<dim3((ROWS*DINNER)/256), 256, 0, stream>>>(
        xz, convw + i*DINNER*4, convb + i*DINNER, uc);

    // x_proj: [32768,512] x [48,512]^T -> xdbl [32768,48]
    k_gemm<2,3,0,false><<<dim3(ROWS/32, 1), 32, 0, stream>>>(
        uc, DINNER, Wx + (size_t)i*48*DINNER, DINNER, xdbl, 48, nullptr, DINNER);

    // dt_proj: [32768,16] x [512,16]^T -> softplus(+bias) -> dtb [32768,512]
    k_gemm<2,4,1,true><<<dim3(ROWS/32, 512/64), 32, 0, stream>>>(
        xdbl, 48, Wdt + (size_t)i*DINNER*DTRANK, DTRANK, dtb, DINNER,
        bdt + i*DINNER, DTRANK);

    k_scan<<<dim3(16), 256, 0, stream>>>(
        uc, dtb, xdbl, xz, Alog + (size_t)i*DINNER*DSTATE, Dp + i*DINNER);

    // out_proj: y [32768,512] (xz cols 0..511) x [256,512]^T, residual into xbuf
    k_gemm<2,4,2,false><<<dim3(ROWS/32, 256/64), 32, 0, stream>>>(
        xz, 1024, Wout + (size_t)i*DMODEL*DINNER, DINNER, xbuf, DMODEL,
        nullptr, DINNER);
  }

  k_lnf_scatter<<<dim3(ROWS/8), 256, 0, stream>>>(xbuf, lnfw, lnfb, co, out);
}
